// NonAutoRegressive_86792699118122
// MI455X (gfx1250) — matrix operand order repaired and checked by reading, not compiled, not run on hardware
//
#include <hip/hip_runtime.h>
#include <hip/hip_bf16.h>

// ---------------------------------------------------------------------------
// Gated-GCN forward for MI455X (gfx1250, wave32, WMMA + TDM).
// Dense 64-dim matmuls -> v_wmma_f32_16x16x32_f16 (K=64 = 2 WMMA steps).
// Weight staging into LDS -> tensor_load_to_lds (Tensor Data Mover, TENSORcnt).
// ---------------------------------------------------------------------------

#define NNODES 20000
#define NEDGES 320000

typedef _Float16     v16h __attribute__((ext_vector_type(16)));
typedef float        v8f  __attribute__((ext_vector_type(8)));
typedef unsigned int v4u  __attribute__((ext_vector_type(4)));
typedef int          v8i  __attribute__((ext_vector_type(8)));
typedef int          v4i  __attribute__((ext_vector_type(4)));

#define DEVINL __device__ __forceinline__

// ---- Tensor Data Mover: 2D f16 tile load Global -> LDS (ISA 8.3/8.4) -------
// rows x cols elements, row stride in elements; one wave issues, EXEC ignored.
DEVINL void tdm_load_2d_f16(unsigned int lds_off, const void* gptr,
                            unsigned int rows, unsigned int cols,
                            unsigned int stride) {
  unsigned long long ga = (unsigned long long)gptr;
  v4u g0;
  g0[0] = 1u;                                   // count=1 (valid user D#)
  g0[1] = lds_off;                              // lds_addr (bytes)
  g0[2] = (unsigned int)(ga & 0xFFFFFFFFu);     // global_addr[31:0]
  g0[3] = (unsigned int)((ga >> 32) & 0x01FFFFFFu) | 0x80000000u; // [56:32]|type=2
  v8i g1;
  g1[0] = (int)(1u << 16);                      // data_size=1 (2 bytes), wg_mask=0
  g1[1] = (int)((cols & 0xFFFFu) << 16);        // tensor_dim0[15:0]
  g1[2] = (int)(((cols >> 16) & 0xFFFFu) | ((rows & 0xFFFFu) << 16)); // dim0 hi | dim1 lo
  g1[3] = (int)(((rows >> 16) & 0xFFFFu) | ((cols & 0xFFFFu) << 16)); // dim1 hi | tile_dim0
  g1[4] = (int)(rows & 0xFFFFu);                // tile_dim1 (tile_dim2 = 0)
  g1[5] = (int)stride;                          // tensor_dim0_stride[31:0]
  g1[6] = 0;                                    // stride hi | dim1_stride lo
  g1[7] = 0;
  v4i gz4 = {0, 0, 0, 0};
  v8i gz8 = {0, 0, 0, 0, 0, 0, 0, 0};
  __builtin_amdgcn_tensor_load_to_lds(g0, g1, gz4, gz4, gz8, 0);
}

DEVINL unsigned int lds_offset(const void* p) {
  // low 32 bits of a flat address that maps to LDS == wave LDS byte address
  return (unsigned int)(unsigned long long)p;
}

// ---- WMMA fragment helpers (m/n lane mapping per CDNA5 ISA 7.12.2; k order contiguous, not the ISA table) ------------------

// A fragment 16x32 f16: row m = lane&15; K contiguous per lane half (k = 16h + s)
DEVINL v16h load_a_frag_row(const _Float16* rowp, int kk, int lane) {
  const _Float16* p = rowp + kk + ((lane >> 4) << 4);
  v16h f;
#pragma unroll
  for (int i = 0; i < 16; ++i) f[i] = p[i];
  return f;
}

DEVINL v16h load_a_frag(const _Float16* base, int ld, int kk, int lane) {
  return load_a_frag_row(base + (lane & 15) * ld, kk, lane);
}

// B fragment 32x16 f16 from WT (row-major [64][64], row = output column)
DEVINL v16h load_b_frag(const _Float16* wt, int n0, int kk, int lane) {
  const _Float16* p = wt + (n0 + (lane & 15)) * 64 + kk + ((lane >> 4) << 4);
  v16h f;
#pragma unroll
  for (int i = 0; i < 16; ++i) f[i] = p[i];
  return f;
}

DEVINL v8f wmma_f16(v16h a, v16h b, v8f c) {
  return __builtin_amdgcn_wmma_f32_16x16x32_f16(
      false, a, false, b, (short)0, c, false, false);
}

// batched 16x64 GEMM step: acc[0..3] += Atile(kk) * WT[:, :] (one K-half)
DEVINL void gemm_khalf(v8f acc[4], v16h a, const _Float16* WT, int kk, int lane) {
  v16h b0 = load_b_frag(WT, 0,  kk, lane);
  v16h b1 = load_b_frag(WT, 16, kk, lane);
  v16h b2 = load_b_frag(WT, 32, kk, lane);
  v16h b3 = load_b_frag(WT, 48, kk, lane);
  acc[0] = wmma_f16(a, b0, acc[0]);
  acc[1] = wmma_f16(a, b1, acc[1]);
  acc[2] = wmma_f16(a, b2, acc[2]);
  acc[3] = wmma_f16(a, b3, acc[3]);
}

// ---- setup kernels ---------------------------------------------------------

__global__ void k_conv(const float* __restrict__ reads,
                       const float* __restrict__ cw, const float* __restrict__ cb,
                       float* __restrict__ h32, _Float16* __restrict__ h16) {
  __shared__ float sr[4][512];
  int node0 = blockIdx.x * 4;
  int tid = threadIdx.x;
  const float* gsrc = reads + (size_t)node0 * 512;
#pragma unroll
  for (int i = 0; i < 8; ++i) {
    int idx = tid + i * 256;
    sr[idx >> 9][idx & 511] = gsrc[idx];
  }
  __syncthreads();
  int ln = tid >> 6;
  int d  = tid & 63;
  float w[12];
#pragma unroll
  for (int j = 0; j < 12; ++j) w[j] = cw[d * 12 + j];
  const float* r0 = sr[ln];
  float mx = -1e30f;
  for (int t = 0; t < 126; ++t) {
    float s = 0.f;
#pragma unroll
    for (int c = 0; c < 4; ++c)
#pragma unroll
      for (int k = 0; k < 3; ++k)
        s += r0[c * 128 + t + k] * w[c * 3 + k];
    mx = fmaxf(mx, s);
  }
  float hv = mx + cb[d];
  int node = node0 + ln;
  h32[node * 64 + d] = hv;
  h16[node * 64 + d] = (_Float16)hv;
}

__global__ void k_encode(const float* __restrict__ ov, const float* __restrict__ ew,
                         const float* __restrict__ eb, float* __restrict__ e32,
                         _Float16* __restrict__ e16) {
  int i = blockIdx.x * 256 + threadIdx.x;
  if (i >= NEDGES * 64) return;
  int c = i & 63;
  float v = ov[i >> 6] * ew[c] + eb[c];
  e32[i] = v;
  e16[i] = (_Float16)v;
}

__global__ void k_zero(float* p, int n) {
  int i = blockIdx.x * 256 + threadIdx.x;
  if (i < n) p[i] = 0.f;
}

struct TPair { const float* src; _Float16* dst; };
struct TList { TPair p[23]; };
__global__ void k_prep(TList tl) {
  const float* s = tl.p[blockIdx.x].src;
  _Float16*    o = tl.p[blockIdx.x].dst;
  for (int i = threadIdx.x; i < 4096; i += 256) {
    int n = i >> 6, k = i & 63;
    o[n * 64 + k] = (_Float16)s[k * 64 + n];
  }
}

// ---- per-layer kernels -----------------------------------------------------

struct NGArgs {
  const _Float16* h16;
  const _Float16 *AT, *BT, *UT, *VT;
  float *HA, *HB, *HU, *HV;
};
__global__ void k_node_gemm(NGArgs a) {
  __shared__ __align__(16) _Float16 W[4][4096];  // 32 KB
  int tid = threadIdx.x;  // 128
  if (tid < 32) {  // wave 0 drives the Tensor Data Mover
    unsigned int base = lds_offset(&W[0][0]);
    tdm_load_2d_f16(base,          a.AT, 64, 64, 64);
    tdm_load_2d_f16(base + 8192,   a.BT, 64, 64, 64);
    tdm_load_2d_f16(base + 16384,  a.UT, 64, 64, 64);
    tdm_load_2d_f16(base + 24576,  a.VT, 64, 64, 64);
    __builtin_amdgcn_s_wait_tensorcnt(0);
  }
  __syncthreads();
  int w = tid >> 5, lane = tid & 31;
  int row0 = (blockIdx.x * 4 + w) * 16;
  if (row0 >= NNODES) return;
  const _Float16* X = a.h16 + (size_t)row0 * 64;
  v16h a0 = load_a_frag(X, 64, 0, lane);
  v16h a1 = load_a_frag(X, 64, 32, lane);
  float* outs[4] = {a.HA, a.HB, a.HU, a.HV};
#pragma unroll
  for (int m = 0; m < 4; ++m) {
    v8f acc[4] = {};
    gemm_khalf(acc, a0, W[m], 0, lane);
    gemm_khalf(acc, a1, W[m], 32, lane);
    float* O = outs[m] + (size_t)row0 * 64;
#pragma unroll
    for (int t = 0; t < 4; ++t)
#pragma unroll
      for (int v = 0; v < 8; ++v)
        O[(((lane >> 4) << 3) + v) * 64 + t * 16 + (lane & 15)] = acc[t][v];
  }
}

struct EArgs {
  const _Float16 *e16, *CT;
  const int *src, *dst;
  const float *HA, *HB, *HV, *b_e;
  float *ehat, *num, *den, *gsum_e, *gsq_e;
};
__global__ void k_edge(EArgs g) {
  __shared__ __align__(16) _Float16 sCT[4096];   // 8 KB
  __shared__ float tile[4][16][66];
  __shared__ float ssum[64], ssq[64];
  int tid = threadIdx.x;  // 128
  if (tid < 32) {
    tdm_load_2d_f16(lds_offset(sCT), g.CT, 64, 64, 64);
    __builtin_amdgcn_s_wait_tensorcnt(0);
  }
  if (tid >= 64 && tid < 128) { ssum[tid - 64] = 0.f; ssq[tid - 64] = 0.f; }
  __syncthreads();
  int w = tid >> 5, lane = tid & 31;
  int ebase = blockIdx.x * 64 + w * 16;
  const _Float16* X = g.e16 + (size_t)ebase * 64;
  __builtin_prefetch(X + 4096, 0, 1);  // next block's e-tile -> global_prefetch_b8
  v16h a0 = load_a_frag(X, 64, 0, lane);
  v16h a1 = load_a_frag(X, 64, 32, lane);
  v8f acc[4] = {};
  gemm_khalf(acc, a0, sCT, 0, lane);
  gemm_khalf(acc, a1, sCT, 32, lane);
#pragma unroll
  for (int t = 0; t < 4; ++t)
#pragma unroll
    for (int v = 0; v < 8; ++v)
      tile[w][((lane >> 4) << 3) + v][t * 16 + (lane & 15)] = acc[t][v];
  __syncthreads();
  int c0 = lane, c1 = lane + 32;
  float be0 = g.b_e[c0], be1 = g.b_e[c1];
  float s0 = 0.f, q0 = 0.f, s1 = 0.f, q1 = 0.f;
  for (int r = 0; r < 16; ++r) {
    int e = ebase + r;
    int sn = g.src[e], dn = g.dst[e];
    size_t so = (size_t)sn * 64, do_ = (size_t)dn * 64;
    float v0 = tile[w][r][c0] + g.HA[so + c0] + g.HB[do_ + c0] + be0;
    float v1 = tile[w][r][c1] + g.HA[so + c1] + g.HB[do_ + c1] + be1;
    g.ehat[(size_t)e * 64 + c0] = v0;
    g.ehat[(size_t)e * 64 + c1] = v1;
    float sg0 = 1.f / (1.f + __expf(-v0));
    float sg1 = 1.f / (1.f + __expf(-v1));
    atomicAdd(&g.num[do_ + c0], sg0 * g.HV[so + c0]);
    atomicAdd(&g.num[do_ + c1], sg1 * g.HV[so + c1]);
    atomicAdd(&g.den[do_ + c0], sg0);
    atomicAdd(&g.den[do_ + c1], sg1);
    s0 += v0; q0 += v0 * v0; s1 += v1; q1 += v1 * v1;
  }
  atomicAdd(&ssum[c0], s0); atomicAdd(&ssq[c0], q0);
  atomicAdd(&ssum[c1], s1); atomicAdd(&ssq[c1], q1);
  __syncthreads();
  if (tid < 64) {
    atomicAdd(&g.gsum_e[tid], ssum[tid]);
    atomicAdd(&g.gsq_e[tid], ssq[tid]);
  }
}

__global__ void k_node_pre(const float* __restrict__ num, const float* __restrict__ den,
                           const float* __restrict__ b_h, float* __restrict__ HU,
                           float* gsum_h, float* gsq_h) {
  __shared__ float ssum[64], ssq[64];
  int tid = threadIdx.x;  // 256
  if (tid < 64) { ssum[tid] = 0.f; ssq[tid] = 0.f; }
  __syncthreads();
  int c = tid & 63, rg = tid >> 6;
  int row0 = blockIdx.x * 32;
  float bh = b_h[c];
  float s = 0.f, q = 0.f;
  for (int i = 0; i < 8; ++i) {
    size_t idx = (size_t)(row0 + rg * 8 + i) * 64 + c;
    float t = HU[idx] + bh + num[idx] / (den[idx] + 1e-6f);
    HU[idx] = t;
    s += t; q += t * t;
  }
  atomicAdd(&ssum[c], s); atomicAdd(&ssq[c], q);
  __syncthreads();
  if (tid < 64) { atomicAdd(&gsum_h[tid], ssum[tid]); atomicAdd(&gsq_h[tid], ssq[tid]); }
}

__global__ void k_stats(const float* stats, const float* g_h, const float* bt_h,
                        const float* g_e, const float* bt_e, float* coef) {
  int tid = threadIdx.x;  // 128
  if (tid < 64) {
    float m = stats[tid] / (float)NNODES;
    float v = stats[64 + tid] / (float)NNODES - m * m;
    float rs = rsqrtf(v + 1e-5f) * g_h[tid];
    coef[tid] = rs;
    coef[64 + tid] = bt_h[tid] - m * rs;
  } else {
    int c = tid - 64;
    float m = stats[128 + c] / (float)NEDGES;
    float v = stats[192 + c] / (float)NEDGES - m * m;
    float rs = rsqrtf(v + 1e-5f) * g_e[c];
    coef[128 + c] = rs;
    coef[192 + c] = bt_e[c] - m * rs;
  }
}

__global__ void k_node_update(const float* __restrict__ t, const float* __restrict__ coef,
                              float* __restrict__ h32, _Float16* __restrict__ h16) {
  int i = blockIdx.x * 256 + threadIdx.x;
  int c = i & 63;
  float v = h32[i] + fmaxf(t[i] * coef[c] + coef[64 + c], 0.f);
  h32[i] = v;
  h16[i] = (_Float16)v;
}

__global__ void k_edge_update(const float* __restrict__ ehat, const float* __restrict__ coef,
                              float* __restrict__ e32, _Float16* __restrict__ e16) {
  int i = blockIdx.x * 256 + threadIdx.x;
  int c = i & 63;
  float v = e32[i] + fmaxf(ehat[i] * coef[128 + c] + coef[192 + c], 0.f);
  e32[i] = v;
  e16[i] = (_Float16)v;
}

// ---- decoder ---------------------------------------------------------------
struct DArgs {
  const _Float16 *h16, *e16, *W1aT, *W1bT, *W1cT;
  const int *src, *dst;
  const float *b1, *w2, *b2;
  float* out;
};
__global__ void k_decode(DArgs g) {
  __shared__ __align__(16) _Float16 Wa[4096], Wb[4096], Wc[4096];  // 24 KB
  __shared__ float ztile[4][16][66];
  __shared__ float w2s[64], b1s[64];
  int tid = threadIdx.x;  // 128
  if (tid < 32) {
    tdm_load_2d_f16(lds_offset(Wa), g.W1aT, 64, 64, 64);
    tdm_load_2d_f16(lds_offset(Wb), g.W1bT, 64, 64, 64);
    tdm_load_2d_f16(lds_offset(Wc), g.W1cT, 64, 64, 64);
    __builtin_amdgcn_s_wait_tensorcnt(0);
  }
  if (tid >= 64 && tid < 128) { w2s[tid - 64] = g.w2[tid - 64]; b1s[tid - 64] = g.b1[tid - 64]; }
  __syncthreads();
  int w = tid >> 5, lane = tid & 31;
  int ebase = blockIdx.x * 64 + w * 16;
  int em = ebase + (lane & 15);
  const _Float16* ps = g.h16 + (size_t)g.src[em] * 64;
  const _Float16* pd = g.h16 + (size_t)g.dst[em] * 64;
  const _Float16* pe = g.e16 + (size_t)em * 64;
  v8f acc[4] = {};
#pragma unroll
  for (int kk2 = 0; kk2 < 2; ++kk2) {
    int kk = kk2 * 32;
    v16h as = load_a_frag_row(ps, kk, lane);
    v16h ad = load_a_frag_row(pd, kk, lane);
    v16h ae = load_a_frag_row(pe, kk, lane);
    gemm_khalf(acc, as, Wa, kk, lane);
    gemm_khalf(acc, ad, Wb, kk, lane);
    gemm_khalf(acc, ae, Wc, kk, lane);
  }
#pragma unroll
  for (int t = 0; t < 4; ++t)
#pragma unroll
    for (int v = 0; v < 8; ++v) {
      int col = t * 16 + (lane & 15);
      ztile[w][((lane >> 4) << 3) + v][col] = fmaxf(acc[t][v] + b1s[col], 0.f);
    }
  __syncthreads();
  int r = lane & 15, half = lane >> 4;
  float p = 0.f;
#pragma unroll
  for (int c = 0; c < 32; ++c) p += ztile[w][r][half * 32 + c] * w2s[half * 32 + c];
  p += __shfl_xor(p, 16, 32);
  if (lane < 16) g.out[ebase + r] = p + g.b2[0];
}

// ---------------------------------------------------------------------------
extern "C" void kernel_launch(void* const* d_in, const int* in_sizes, int n_in,
                              void* d_out, int out_size, void* d_ws, size_t ws_size,
                              hipStream_t stream) {
  const int N = NNODES, E = NEDGES;

  const float *reads, *overlap, *conv_w, *conv_b, *enc_w, *enc_b;
  const float *dec_w1, *dec_b1, *dec_w2, *dec_b2;
  const int *src, *dst;
  const float *LA[4], *LB[4], *LC[4], *LU[4], *LV[4];
  const float *Lbe[4], *Lbh[4], *Lbte[4], *Lbth[4], *Lge[4], *Lgh[4];

  if (n_in >= 56 && in_sizes[0] == E) {
    // jax.tree_util order (dict keys sorted at every level)
    dst     = (const int*)d_in[0];
    overlap = (const float*)d_in[1];
    conv_b  = (const float*)d_in[2];
    conv_w  = (const float*)d_in[3];
    dec_b1  = (const float*)d_in[4];
    dec_b2  = (const float*)d_in[5];
    dec_w1  = (const float*)d_in[6];
    dec_w2  = (const float*)d_in[7];
    enc_b   = (const float*)d_in[8];
    enc_w   = (const float*)d_in[9];
    for (int l = 0; l < 4; ++l) {
      int b = 10 + l * 11;  // A,B,C,U,V,b_e,b_h,bt_e,bt_h,g_e,g_h
      LA[l]  = (const float*)d_in[b + 0];
      LB[l]  = (const float*)d_in[b + 1];
      LC[l]  = (const float*)d_in[b + 2];
      LU[l]  = (const float*)d_in[b + 3];
      LV[l]  = (const float*)d_in[b + 4];
      Lbe[l] = (const float*)d_in[b + 5];
      Lbh[l] = (const float*)d_in[b + 6];
      Lbte[l]= (const float*)d_in[b + 7];
      Lbth[l]= (const float*)d_in[b + 8];
      Lge[l] = (const float*)d_in[b + 9];
      Lgh[l] = (const float*)d_in[b + 10];
    }
    reads = (const float*)d_in[54];
    src   = (const int*)d_in[55];
  } else {
    // insertion order
    reads   = (const float*)d_in[0];
    overlap = (const float*)d_in[1];
    src     = (const int*)d_in[2];
    dst     = (const int*)d_in[3];
    conv_w  = (const float*)d_in[4];
    conv_b  = (const float*)d_in[5];
    enc_w   = (const float*)d_in[6];
    enc_b   = (const float*)d_in[7];
    for (int l = 0; l < 4; ++l) {
      int b = 8 + l * 11;  // U,V,A,B,C,b_e,b_h,g_h,bt_h,g_e,bt_e
      LU[l]  = (const float*)d_in[b + 0];
      LV[l]  = (const float*)d_in[b + 1];
      LA[l]  = (const float*)d_in[b + 2];
      LB[l]  = (const float*)d_in[b + 3];
      LC[l]  = (const float*)d_in[b + 4];
      Lbe[l] = (const float*)d_in[b + 5];
      Lbh[l] = (const float*)d_in[b + 6];
      Lgh[l] = (const float*)d_in[b + 7];
      Lbth[l]= (const float*)d_in[b + 8];
      Lge[l] = (const float*)d_in[b + 9];
      Lbte[l]= (const float*)d_in[b + 10];
    }
    dec_w1 = (const float*)d_in[52];
    dec_b1 = (const float*)d_in[53];
    dec_w2 = (const float*)d_in[54];
    dec_b2 = (const float*)d_in[55];
  }

  // ---- workspace carve (fully rewritten every call) ----
  char* wp = (char*)d_ws;
  auto take = [&](size_t bytes) -> void* {
    void* p = (void*)wp;
    wp += (bytes + 255) & ~(size_t)255;
    return p;
  };
  float*     h32  = (float*)take((size_t)N * 64 * 4);
  _Float16*  h16  = (_Float16*)take((size_t)N * 64 * 2);
  float*     e32  = (float*)take((size_t)E * 64 * 4);
  _Float16*  e16  = (_Float16*)take((size_t)E * 64 * 2);
  float*     ehat = (float*)take((size_t)E * 64 * 4);
  float*     HA   = (float*)take((size_t)N * 64 * 4);
  float*     HB   = (float*)take((size_t)N * 64 * 4);
  float*     HU   = (float*)take((size_t)N * 64 * 4);
  float*     HV   = (float*)take((size_t)N * 64 * 4);
  float*     num  = (float*)take((size_t)N * 64 * 4);
  float*     den  = (float*)take((size_t)N * 64 * 4);
  float*     stats= (float*)take(256 * 4);
  float*     coef = (float*)take(256 * 4);
  _Float16*  wt16 = (_Float16*)take((size_t)23 * 4096 * 2);
  if ((size_t)(wp - (char*)d_ws) > ws_size) return;

  // ---- weight prep: f32 -> f16 transposed ----
  TList tl;
  for (int l = 0; l < 4; ++l) {
    const float* mats[5] = {LA[l], LB[l], LC[l], LU[l], LV[l]};
    for (int j = 0; j < 5; ++j) {
      tl.p[l * 5 + j].src = mats[j];
      tl.p[l * 5 + j].dst = wt16 + (size_t)(l * 5 + j) * 4096;
    }
  }
  for (int jb = 0; jb < 3; ++jb) {
    tl.p[20 + jb].src = dec_w1 + (size_t)jb * 4096;
    tl.p[20 + jb].dst = wt16 + (size_t)(20 + jb) * 4096;
  }
  k_prep<<<23, 256, 0, stream>>>(tl);

  // ---- embedders ----
  k_conv<<<N / 4, 256, 0, stream>>>(reads, conv_w, conv_b, h32, h16);
  k_encode<<<(E * 64) / 256, 256, 0, stream>>>(overlap, enc_w, enc_b, e32, e16);

  // ---- GCN layers ----
  const int nzero = 2 * N * 64 + 256;  // num, den, stats (contiguous)
  for (int l = 0; l < 4; ++l) {
    const _Float16* base = wt16 + (size_t)l * 5 * 4096;
    k_zero<<<(nzero + 255) / 256, 256, 0, stream>>>(num, nzero);

    NGArgs ng;
    ng.h16 = h16;
    ng.AT = base; ng.BT = base + 4096; ng.UT = base + 3 * 4096; ng.VT = base + 4 * 4096;
    ng.HA = HA; ng.HB = HB; ng.HU = HU; ng.HV = HV;
    k_node_gemm<<<(N / 16 + 3) / 4, 128, 0, stream>>>(ng);

    EArgs ea;
    ea.e16 = e16; ea.CT = base + 2 * 4096;
    ea.src = src; ea.dst = dst;
    ea.HA = HA; ea.HB = HB; ea.HV = HV; ea.b_e = Lbe[l];
    ea.ehat = ehat; ea.num = num; ea.den = den;
    ea.gsum_e = stats + 128; ea.gsq_e = stats + 192;
    k_edge<<<E / 64, 128, 0, stream>>>(ea);

    k_node_pre<<<N / 32, 256, 0, stream>>>(num, den, Lbh[l], HU, stats, stats + 64);
    k_stats<<<1, 128, 0, stream>>>(stats, Lgh[l], Lbth[l], Lge[l], Lbte[l], coef);
    k_node_update<<<(N * 64) / 256, 256, 0, stream>>>(HU, coef, h32, h16);
    k_edge_update<<<(E * 64) / 256, 256, 0, stream>>>(ehat, coef, e32, e16);
  }

  // ---- decoder ----
  DArgs da;
  da.h16 = h16; da.e16 = e16;
  da.W1aT = wt16 + (size_t)20 * 4096;
  da.W1bT = wt16 + (size_t)21 * 4096;
  da.W1cT = wt16 + (size_t)22 * 4096;
  da.src = src; da.dst = dst;
  da.b1 = dec_b1; da.w2 = dec_w2; da.b2 = dec_b2;
  da.out = (float*)d_out;
  k_decode<<<E / 64, 128, 0, stream>>>(da);
}
